// CausalSelfAttention_52888227283716
// MI455X (gfx1250) — matrix operand order repaired and checked by reading, not compiled, not run on hardware
//
#include <hip/hip_runtime.h>
#include <hip/hip_bf16.h>

#define B_  2
#define T_  2048
#define D_  1024
#define H_  16
#define HD_ 64
#define M_  (B_*T_)   // 4096 rows for the big GEMMs

typedef __attribute__((ext_vector_type(16))) __bf16 v16bf;
typedef __attribute__((ext_vector_type(8)))  __bf16 v8bf;
typedef __attribute__((ext_vector_type(8)))  float  v8f;

static __device__ __forceinline__ v8f wmma_bf16(v16bf a, v16bf b, v8f c) {
  // D = A(16x32 bf16) x B(32x16 bf16) + C(16x16 f32)
  return __builtin_amdgcn_wmma_f32_16x16x32_bf16(false, a, false, b, (short)0, c,
                                                 false, false);
}

// Build a v16bf A-operand from two contiguous 8-halfword chunks
// (A layout: halfs 0..15 = K base..base+15, contiguous).
static __device__ __forceinline__ v16bf load_a16(const __bf16* p0, const __bf16* p1) {
  v8bf lo = *(const v8bf*)p0;
  v8bf hi = *(const v8bf*)p1;
  v16bf r;
#pragma unroll
  for (int i = 0; i < 8; ++i) { r[i] = lo[i]; r[8 + i] = hi[i]; }
  return r;
}

__global__ void cvt_f32_bf16(const float* __restrict__ in, __bf16* __restrict__ out, int n) {
  int i = blockIdx.x * blockDim.x + threadIdx.x;
  if (i < n) out[i] = (__bf16)in[i];
}

// y[m,n] = sum_k A[m,k] * W[n,k]   (A: [4096,1024] bf16, W: [1024,1024] bf16 row-major)
// Each wave computes a 16x64 tile: mt in [0,256), nc in [0,16).
// mode 0: Q -> [B,H,T,HD] bf16, scaled by 1/sqrt(HD)
// mode 1: K -> [B,H,T,HD] bf16
// mode 2: V -> [B,H,HD,T] bf16 (transposed for attention B-operand)
// mode 3: final projection -> [B,T,D] f32
__global__ void gemm_bf16(const __bf16* __restrict__ A, const __bf16* __restrict__ W,
                          void* __restrict__ out, int mode) {
  int lane = threadIdx.x & 31;
  int wid  = blockIdx.x * (blockDim.x >> 5) + (threadIdx.x >> 5);
  int mt   = wid & 255;
  int nc   = wid >> 8;
  int col  = lane & 15;
  int hi   = lane >> 4;

  const __bf16* arow = A + (size_t)(mt * 16 + col) * D_;
  v8f acc[4] = {};
  for (int kk = 0; kk < D_; kk += 32) {
    v16bf a = load_a16(arow + kk + hi * 16, arow + kk + hi * 16 + 8);
#pragma unroll
    for (int j = 0; j < 4; ++j) {
      int n = nc * 64 + j * 16 + col;   // B-operand: n = lane&15, K contiguous
      v16bf bo = *(const v16bf*)(W + (size_t)n * D_ + kk + hi * 16);
      acc[j] = wmma_bf16(a, bo, acc[j]);
    }
  }

  if (mode == 3) {
    float* o = (float*)out;
#pragma unroll
    for (int r = 0; r < 8; ++r) {
      int m = mt * 16 + r + 8 * hi;     // C/D layout: row = r + 8*(lane>=16)
#pragma unroll
      for (int j = 0; j < 4; ++j) {
        int n = nc * 64 + j * 16 + col;
        o[(size_t)m * D_ + n] = acc[j][r];
      }
    }
  } else {
    __bf16* o = (__bf16*)out;
    float s = (mode == 0) ? 0.125f : 1.0f;   // 1/sqrt(64) folded into Q
#pragma unroll
    for (int r = 0; r < 8; ++r) {
      int m  = mt * 16 + r + 8 * hi;
      int bb = m >> 11;
      int t  = m & (T_ - 1);
#pragma unroll
      for (int j = 0; j < 4; ++j) {
        int n  = nc * 64 + j * 16 + col;
        int hh = n >> 6;
        int d  = n & 63;
        size_t idx = (mode == 2)
            ? ((size_t)(bb * H_ + hh) * HD_ + d) * T_ + t     // V transposed
            : ((size_t)(bb * H_ + hh) * T_ + t) * HD_ + d;    // Q/K head-major
        o[idx] = (__bf16)(acc[j][r] * s);
      }
    }
  }
}

// Flash attention: one wave per (b, h, 16-query tile). Key loop in 32-key pairs.
__global__ void attn_fwd(const __bf16* __restrict__ Q, const __bf16* __restrict__ K,
                         const __bf16* __restrict__ Vt, __bf16* __restrict__ O) {
  __shared__ __bf16 plds[4][16 * 32];
  int lane = threadIdx.x & 31;
  int wv   = threadIdx.x >> 5;
  int wid  = blockIdx.x * 4 + wv;
  int b    = wid >> 11;
  int h    = (wid >> 7) & 15;
  int qt   = wid & 127;
  int col  = lane & 15;
  int hi   = lane >> 4;

  const __bf16* qrow = Q + ((size_t)(b * H_ + h) * T_ + qt * 16 + col) * HD_;
  v16bf a0 = load_a16(qrow + hi * 16,      qrow + hi * 16 + 8);   // d = 0..31
  v16bf a1 = load_a16(qrow + 32 + hi * 16, qrow + 40 + hi * 16);  // d = 32..63
  const __bf16* Kb = K  + (size_t)(b * H_ + h) * T_ * HD_;
  const __bf16* Vb = Vt + (size_t)(b * H_ + h) * HD_ * T_;
  __bf16* pl = plds[wv];

  float mrow[8], lrow[8];
#pragma unroll
  for (int r = 0; r < 8; ++r) { mrow[r] = -1e30f; lrow[r] = 0.0f; }
  v8f acc[4] = {};

  int npair = (qt * 16 + 15) / 32 + 1;
  for (int kp = 0; kp < npair; ++kp) {
    int k0 = kp * 32;
    // scores: S = Q(16x64) @ K^T -> two 16x16 accumulators (keys k0..+15, k0+16..+31)
    v8f s0 = {}, s1 = {};
    const __bf16* kr0 = Kb + (size_t)(k0 + col) * HD_;
    s0 = wmma_bf16(a0, *(const v16bf*)(kr0 + hi * 16),      s0);
    s0 = wmma_bf16(a1, *(const v16bf*)(kr0 + 32 + hi * 16), s0);
    const __bf16* kr1 = Kb + (size_t)(k0 + 16 + col) * HD_;
    s1 = wmma_bf16(a0, *(const v16bf*)(kr1 + hi * 16),      s1);
    s1 = wmma_bf16(a1, *(const v16bf*)(kr1 + 32 + hi * 16), s1);

#pragma unroll
    for (int r = 0; r < 8; ++r) {
      int row = qt * 16 + r + 8 * hi;
      float v0 = (k0 + col      <= row) ? s0[r] : -1e30f;   // causal mask
      float v1 = (k0 + 16 + col <= row) ? s1[r] : -1e30f;
      float mx = fmaxf(v0, v1);
#pragma unroll
      for (int m = 1; m <= 8; m <<= 1)        // reduce within 16-lane half
        mx = fmaxf(mx, __shfl_xor(mx, m, 32));
      float mn   = fmaxf(mrow[r], mx);
      float corr = __expf(mrow[r] - mn);
      float p0   = __expf(v0 - mn);
      float p1   = __expf(v1 - mn);
      float rs   = p0 + p1;
#pragma unroll
      for (int m = 1; m <= 8; m <<= 1)
        rs += __shfl_xor(rs, m, 32);
      lrow[r] = lrow[r] * corr + rs;
      mrow[r] = mn;
#pragma unroll
      for (int c = 0; c < 4; ++c) acc[c][r] *= corr;
      // stash P in C/D layout -> LDS (row-major 16x32 bf16)
      int rowL = r + 8 * hi;
      pl[rowL * 32 + col]      = (__bf16)p0;
      pl[rowL * 32 + 16 + col] = (__bf16)p1;
    }
    // same-wave LDS RAW: DS is in-order per wave, but keep compiler honest
    asm volatile("s_wait_dscnt 0" ::: "memory");

    // reload P in A-operand layout (two contiguous 8-half chunks per lane)
    const __bf16* prow = pl + col * 32;
    v16bf pa = load_a16(prow + hi * 16, prow + hi * 16 + 8);

    // O_tile += P(16x32) @ V(32x64); Vt rows give contiguous B-operand loads
#pragma unroll
    for (int c = 0; c < 4; ++c) {
      const __bf16* vr = Vb + (size_t)(c * 16 + col) * T_ + k0 + hi * 16;
      acc[c] = wmma_bf16(pa, *(const v16bf*)vr, acc[c]);
    }
  }

#pragma unroll
  for (int r = 0; r < 8; ++r) {
    float inv = 1.0f / lrow[r];
    int t = qt * 16 + r + 8 * hi;
#pragma unroll
    for (int c = 0; c < 4; ++c) {
      int d = c * 16 + col;
      O[(size_t)(b * T_ + t) * D_ + h * HD_ + d] = (__bf16)(acc[c][r] * inv);
    }
  }
}

extern "C" void kernel_launch(void* const* d_in, const int* in_sizes, int n_in,
                              void* d_out, int out_size, void* d_ws, size_t ws_size,
                              hipStream_t stream) {
  const float* x  = (const float*)d_in[0];
  const float* Wq = (const float*)d_in[1];
  const float* Wk = (const float*)d_in[2];
  const float* Wv = (const float*)d_in[3];
  const float* Wo = (const float*)d_in[4];
  // d_in[5] = causal mask: statically known, ignored.
  float* out = (float*)d_out;

  char* ws = (char*)d_ws;                       // 48 MB layout:
  __bf16* xb  = (__bf16*)(ws);                  //  0 +  8 MB  x (bf16)
  __bf16* wqb = (__bf16*)(ws + ( 8u << 20));    //  8 +  2 MB
  __bf16* wkb = (__bf16*)(ws + (10u << 20));    // 10 +  2 MB
  __bf16* wvb = (__bf16*)(ws + (12u << 20));    // 12 +  2 MB
  __bf16* wob = (__bf16*)(ws + (14u << 20));    // 14 +  2 MB
  __bf16* qh  = (__bf16*)(ws + (16u << 20));    // 16 +  8 MB  Q [B,H,T,HD]
  __bf16* kh  = (__bf16*)(ws + (24u << 20));    // 24 +  8 MB  K [B,H,T,HD]
  __bf16* vth = (__bf16*)(ws + (32u << 20));    // 32 +  8 MB  V [B,H,HD,T]
  __bf16* ob  = (__bf16*)(ws + (40u << 20));    // 40 +  8 MB  attn out [B,T,D]

  const int nx = M_ * D_;     // 4M
  const int nw = D_ * D_;     // 1M
  cvt_f32_bf16<<<(nx + 255) / 256, 256, 0, stream>>>(x,  xb,  nx);
  cvt_f32_bf16<<<(nw + 255) / 256, 256, 0, stream>>>(Wq, wqb, nw);
  cvt_f32_bf16<<<(nw + 255) / 256, 256, 0, stream>>>(Wk, wkb, nw);
  cvt_f32_bf16<<<(nw + 255) / 256, 256, 0, stream>>>(Wv, wvb, nw);
  cvt_f32_bf16<<<(nw + 255) / 256, 256, 0, stream>>>(Wo, wob, nw);

  // QKV projections: 4096 waves each (256 m-tiles x 16 n-chunks), 4 waves/block
  gemm_bf16<<<1024, 128, 0, stream>>>(xb, wqb, (void*)qh,  0);
  gemm_bf16<<<1024, 128, 0, stream>>>(xb, wkb, (void*)kh,  1);
  gemm_bf16<<<1024, 128, 0, stream>>>(xb, wvb, (void*)vth, 2);

  // attention: 4096 waves = 2*16*128 (b,h,query-tile)
  attn_fwd<<<1024, 128, 0, stream>>>(qh, kh, vth, ob);

  // output projection -> f32 d_out
  gemm_bf16<<<1024, 128, 0, stream>>>(ob, wob, (void*)out, 3);
}
